// FusedWhisperLowBitEncoderlayer_6777458393343
// MI455X (gfx1250) — matrix operand order repaired and checked by reading, not compiled, not run on hardware
//
#include <hip/hip_runtime.h>
#include <hip/hip_bf16.h>
#include <math.h>

// Problem constants (match reference)
#define S_LEN   1500
#define DMODEL  1280
#define NHEAD   20
#define HDIM    64
#define FFN_DIM 5120
#define MPAD    1536   // S padded to multiple of 64 for GEMM tiling

typedef __attribute__((ext_vector_type(16))) __bf16 v16bf;
typedef __attribute__((ext_vector_type(8)))  __bf16 v8bf;
typedef __attribute__((ext_vector_type(8)))  float  v8f;

static __device__ __forceinline__ unsigned short f2bf_bits(float f) {
    unsigned int u = __float_as_uint(f);
    unsigned int r = (u + 0x7FFFu + ((u >> 16) & 1u)) >> 16;  // RNE
    return (unsigned short)r;
}

// ---------------------------------------------------------------------------
// fp32 -> bf16 conversion (weights)
// ---------------------------------------------------------------------------
__global__ void convert_f32_to_bf16(const float* __restrict__ src,
                                    unsigned short* __restrict__ dst, int n) {
    int i = blockIdx.x * blockDim.x + threadIdx.x;
    if (i < n) dst[i] = f2bf_bits(src[i]);
}

// ---------------------------------------------------------------------------
// LayerNorm (one 256-thread block per row), emits bf16; rows >= S zeroed
// ---------------------------------------------------------------------------
__global__ __launch_bounds__(256) void layernorm_to_bf16(
    const float* __restrict__ x, const float* __restrict__ w,
    const float* __restrict__ b, unsigned short* __restrict__ out) {
    int row = blockIdx.x;
    int tid = threadIdx.x;
    __shared__ float red[256];
    if (row >= S_LEN) {
        for (int c = tid; c < DMODEL; c += 256) out[row * DMODEL + c] = 0;
        return;
    }
    const float* xr = x + (long)row * DMODEL;
    float local = 0.f;
    for (int c = tid; c < DMODEL; c += 256) local += xr[c];
    red[tid] = local; __syncthreads();
    for (int off = 128; off > 0; off >>= 1) { if (tid < off) red[tid] += red[tid + off]; __syncthreads(); }
    float mean = red[0] * (1.0f / DMODEL);
    __syncthreads();
    local = 0.f;
    for (int c = tid; c < DMODEL; c += 256) { float d = xr[c] - mean; local += d * d; }
    red[tid] = local; __syncthreads();
    for (int off = 128; off > 0; off >>= 1) { if (tid < off) red[tid] += red[tid + off]; __syncthreads(); }
    float rstd = rsqrtf(red[0] * (1.0f / DMODEL) + 1e-5f);
    __syncthreads();
    for (int c = tid; c < DMODEL; c += 256) {
        float v = w[c] * ((xr[c] - mean) * rstd) + b[c];
        out[row * DMODEL + c] = f2bf_bits(v);
    }
}

// ---------------------------------------------------------------------------
// WMMA GEMM: out[m,n] = epi( alpha*(sum_k A[m,k]*W[n,k] + bias[n]) ) (+res)
// A: bf16 [MPAD, K] row-major;  W: bf16 [N, K] row-major (i.e. B = W^T)
// Each wave computes a 16x64 tile; 4 waves/block -> 64x64 per block.
// ---------------------------------------------------------------------------
__global__ __launch_bounds__(128) void gemm_wmma_bf16(
    const unsigned short* __restrict__ A,
    const unsigned short* __restrict__ W,
    const float* __restrict__ bias,       // [N] or nullptr
    const float* __restrict__ residual,   // [resRows, N] or nullptr
    float* __restrict__ outF,             // or nullptr; rows < storeRowsF
    unsigned short* __restrict__ outB,    // or nullptr; all MPAD rows
    int K, int N, float alpha, int gelu, int resRows, int storeRowsF)
{
    int lane = threadIdx.x & 31;
    int wave = threadIdx.x >> 5;
    int g    = lane >> 4;
    int l16  = lane & 15;
    int m0   = blockIdx.x * 64 + wave * 16;
    int n0   = blockIdx.y * 64;

    v8f acc[4];
    #pragma unroll
    for (int nb = 0; nb < 4; nb++)
        #pragma unroll
        for (int r = 0; r < 8; r++) acc[nb][r] = 0.f;

    const unsigned short* arow = A + (long)(m0 + l16) * K;
    for (int k0 = 0; k0 < K; k0 += 32) {
        v16bf a;
        {
            v8bf lo = *(const v8bf*)(arow + k0 + g * 8);
            v8bf hi = *(const v8bf*)(arow + k0 + 16 + g * 8);
            #pragma unroll
            for (int j = 0; j < 8; j++) { a[j] = lo[j]; a[8 + j] = hi[j]; }
        }
        #pragma unroll
        for (int nb = 0; nb < 4; nb++) {
            const unsigned short* wr = W + (long)(n0 + nb * 16 + l16) * K + k0;
            v8bf blo = *(const v8bf*)(wr + g * 8);
            v8bf bhi = *(const v8bf*)(wr + 16 + g * 8);
            v16bf bfrag;
            #pragma unroll
            for (int j = 0; j < 8; j++) { bfrag[j] = blo[j]; bfrag[8 + j] = bhi[j]; }
            acc[nb] = __builtin_amdgcn_wmma_f32_16x16x32_bf16(
                false, a, false, bfrag, (short)0, acc[nb], false, false);
        }
    }

    #pragma unroll
    for (int nb = 0; nb < 4; nb++) {
        int col = n0 + nb * 16 + l16;
        float bv = bias ? bias[col] : 0.f;
        #pragma unroll
        for (int r = 0; r < 8; r++) {
            int row = m0 + r + g * 8;
            float v = alpha * (acc[nb][r] + bv);
            if (gelu) v = 0.5f * v * (1.f + erff(v * 0.7071067811865475f));
            if (residual && row < resRows) v += residual[(long)row * N + col];
            if (outF && row < storeRowsF) outF[(long)row * N + col] = v;
            if (outB) outB[(long)row * N + col] = f2bf_bits(v);
        }
    }
}

// ---------------------------------------------------------------------------
// Flash attention: one wave per (16-query tile, head). Online softmax.
// Q pre-scaled by 1/sqrt(HD) in the Q GEMM epilogue.
// ---------------------------------------------------------------------------
__global__ __launch_bounds__(32) void flash_attn_wmma(
    const unsigned short* __restrict__ Q, const unsigned short* __restrict__ Kb,
    const unsigned short* __restrict__ V, unsigned short* __restrict__ O)
{
    int lane  = threadIdx.x & 31;
    int g     = lane >> 4;
    int l16   = lane & 15;
    int qrow0 = blockIdx.x * 16;
    int cbase = blockIdx.y * HDIM;

    __shared__ __align__(16) float          sS[16][32];
    __shared__ __align__(16) unsigned short sP[16][32];
    __shared__ float sF[16];
    __shared__ float sInv[16];

    // Q fragments: features [0,32) and [32,64)
    v16bf qa0, qa1;
    {
        const unsigned short* qr = Q + (long)(qrow0 + l16) * DMODEL + cbase;
        v8bf lo0 = *(const v8bf*)(qr + g * 8);
        v8bf hi0 = *(const v8bf*)(qr + 16 + g * 8);
        v8bf lo1 = *(const v8bf*)(qr + 32 + g * 8);
        v8bf hi1 = *(const v8bf*)(qr + 48 + g * 8);
        #pragma unroll
        for (int j = 0; j < 8; j++) {
            qa0[j] = lo0[j]; qa0[8 + j] = hi0[j];
            qa1[j] = lo1[j]; qa1[8 + j] = hi1[j];
        }
    }

    v8f acc[4];
    #pragma unroll
    for (int nb = 0; nb < 4; nb++)
        #pragma unroll
        for (int r = 0; r < 8; r++) acc[nb][r] = 0.f;
    float m_i = -1e30f, l_i = 0.f;

    const int NKT = (S_LEN + 31) / 32;   // 47 key tiles of 32
    for (int kt = 0; kt < NKT; kt++) {
        int kbase = kt * 32;
        // scores S = Q * K^T for 32 keys
        #pragma unroll
        for (int nb = 0; nb < 2; nb++) {
            const unsigned short* kr = Kb + (long)(kbase + nb * 16 + l16) * DMODEL + cbase;
            v8bf bl0 = *(const v8bf*)(kr + g * 8);
            v8bf bh0 = *(const v8bf*)(kr + 16 + g * 8);
            v8bf bl1 = *(const v8bf*)(kr + 32 + g * 8);
            v8bf bh1 = *(const v8bf*)(kr + 48 + g * 8);
            v16bf b0, b1;
            #pragma unroll
            for (int j = 0; j < 8; j++) {
                b0[j] = bl0[j]; b0[8 + j] = bh0[j];
                b1[j] = bl1[j]; b1[8 + j] = bh1[j];
            }
            v8f c;
            #pragma unroll
            for (int r = 0; r < 8; r++) c[r] = 0.f;
            c = __builtin_amdgcn_wmma_f32_16x16x32_bf16(false, qa0, false, b0, (short)0, c, false, false);
            c = __builtin_amdgcn_wmma_f32_16x16x32_bf16(false, qa1, false, b1, (short)0, c, false, false);
            #pragma unroll
            for (int r = 0; r < 8; r++) sS[r + g * 8][nb * 16 + l16] = c[r];
        }
        __syncthreads();

        if (lane < 16) {
            int m = lane;
            float mmax = m_i;
            float sc[32];
            #pragma unroll
            for (int j = 0; j < 32; j++) {
                float s = sS[m][j];
                if (kbase + j >= S_LEN) s = -1e30f;
                sc[j] = s;
                mmax = fmaxf(mmax, s);
            }
            float f = __expf(m_i - mmax);
            float lsum = 0.f;
            #pragma unroll
            for (int j = 0; j < 32; j++) {
                float p = __expf(sc[j] - mmax);
                lsum += p;
                sP[m][j] = f2bf_bits(p);
            }
            l_i = l_i * f + lsum;
            m_i = mmax;
            sF[m] = f;
        }
        __syncthreads();

        // rescale accumulators by per-row factor
        float fr[8];
        #pragma unroll
        for (int r = 0; r < 8; r++) fr[r] = sF[r + g * 8];
        #pragma unroll
        for (int nb = 0; nb < 4; nb++)
            #pragma unroll
            for (int r = 0; r < 8; r++) acc[nb][r] *= fr[r];

        // P fragment from LDS (A: 16x32)
        v16bf pa;
        {
            const unsigned short* pr = &sP[l16][0];
            v8bf lo = *(const v8bf*)(pr + g * 8);
            v8bf hi = *(const v8bf*)(pr + 16 + g * 8);
            #pragma unroll
            for (int j = 0; j < 8; j++) { pa[j] = lo[j]; pa[8 + j] = hi[j]; }
        }
        // ctx += P * V  (V B-fragment: B[k,n] = V[kbase+k][cbase+nb*16+n])
        #pragma unroll
        for (int nb = 0; nb < 4; nb++) {
            v16bf vb;
            #pragma unroll
            for (int j = 0; j < 16; j++)
                vb[j] = ((const __bf16*)V)[(long)(kbase + (j < 8 ? g * 8 + j : 16 + g * 8 + (j - 8))) * DMODEL + cbase + nb * 16 + l16];
            acc[nb] = __builtin_amdgcn_wmma_f32_16x16x32_bf16(
                false, pa, false, vb, (short)0, acc[nb], false, false);
        }
        __syncthreads();
    }

    if (lane < 16) sInv[lane] = 1.f / l_i;
    __syncthreads();
    float inv[8];
    #pragma unroll
    for (int r = 0; r < 8; r++) inv[r] = sInv[r + g * 8];
    #pragma unroll
    for (int nb = 0; nb < 4; nb++)
        #pragma unroll
        for (int r = 0; r < 8; r++)
            O[(long)(qrow0 + r + g * 8) * DMODEL + cbase + nb * 16 + l16] =
                f2bf_bits(acc[nb][r] * inv[r]);
}

// ---------------------------------------------------------------------------
// Host launcher
// ---------------------------------------------------------------------------
extern "C" void kernel_launch(void* const* d_in, const int* in_sizes, int n_in,
                              void* d_out, int out_size, void* d_ws, size_t ws_size,
                              hipStream_t stream) {
    const float* x       = (const float*)d_in[0];
    const float* ln1_w   = (const float*)d_in[1];
    const float* ln1_b   = (const float*)d_in[2];
    const float* ln2_w   = (const float*)d_in[3];
    const float* ln2_b   = (const float*)d_in[4];
    const float* wq      = (const float*)d_in[5];
    const float* q_bias  = (const float*)d_in[6];
    const float* wk      = (const float*)d_in[7];
    const float* wv      = (const float*)d_in[8];
    const float* v_bias  = (const float*)d_in[9];
    const float* wo      = (const float*)d_in[10];
    const float* out_bias= (const float*)d_in[11];
    const float* w_fc1   = (const float*)d_in[12];
    const float* fc1_bias= (const float*)d_in[13];
    const float* w_fc2   = (const float*)d_in[14];
    const float* fc2_bias= (const float*)d_in[15];
    float* out = (float*)d_out;

    // workspace carve-up (bytes, 256-aligned sections)
    char* ws = (char*)d_ws;
    const size_t SZ_DD  = (size_t)DMODEL * DMODEL * 2;
    const size_t SZ_FD  = (size_t)FFN_DIM * DMODEL * 2;
    const size_t SZ_ACT = (size_t)MPAD * DMODEL * 2;
    const size_t SZ_ACTF= (size_t)MPAD * DMODEL * 4;
    const size_t SZ_FC1 = (size_t)MPAD * FFN_DIM * 2;
    size_t off = 0;
    unsigned short* wq_bf  = (unsigned short*)(ws + off); off += SZ_DD;
    unsigned short* wk_bf  = (unsigned short*)(ws + off); off += SZ_DD;
    unsigned short* wv_bf  = (unsigned short*)(ws + off); off += SZ_DD;
    unsigned short* wo_bf  = (unsigned short*)(ws + off); off += SZ_DD;
    unsigned short* wf1_bf = (unsigned short*)(ws + off); off += SZ_FD;
    unsigned short* wf2_bf = (unsigned short*)(ws + off); off += SZ_FD;
    unsigned short* h_bf   = (unsigned short*)(ws + off); off += SZ_ACT;
    unsigned short* q_bf   = (unsigned short*)(ws + off); off += SZ_ACT;
    unsigned short* k_bf   = (unsigned short*)(ws + off); off += SZ_ACT;
    unsigned short* v_bf   = (unsigned short*)(ws + off); off += SZ_ACT;
    unsigned short* ctx_bf = (unsigned short*)(ws + off); off += SZ_ACT;
    float*          hres   = (float*)(ws + off);          off += SZ_ACTF;
    unsigned short* h2_bf  = (unsigned short*)(ws + off); off += SZ_ACT;
    unsigned short* fc1_bf = (unsigned short*)(ws + off); off += SZ_FC1;
    (void)ws_size; (void)in_sizes; (void)n_in; (void)out_size;

    const int nDD = DMODEL * DMODEL;
    const int nFD = FFN_DIM * DMODEL;
    convert_f32_to_bf16<<<(nDD + 255) / 256, 256, 0, stream>>>(wq, wq_bf, nDD);
    convert_f32_to_bf16<<<(nDD + 255) / 256, 256, 0, stream>>>(wk, wk_bf, nDD);
    convert_f32_to_bf16<<<(nDD + 255) / 256, 256, 0, stream>>>(wv, wv_bf, nDD);
    convert_f32_to_bf16<<<(nDD + 255) / 256, 256, 0, stream>>>(wo, wo_bf, nDD);
    convert_f32_to_bf16<<<(nFD + 255) / 256, 256, 0, stream>>>(w_fc1, wf1_bf, nFD);
    convert_f32_to_bf16<<<(nFD + 255) / 256, 256, 0, stream>>>(w_fc2, wf2_bf, nFD);

    // LN1
    layernorm_to_bf16<<<MPAD, 256, 0, stream>>>(x, ln1_w, ln1_b, h_bf);

    dim3 blk(128);
    dim3 gD(MPAD / 64, DMODEL / 64);   // 24 x 20
    dim3 gF(MPAD / 64, FFN_DIM / 64);  // 24 x 80
    const float scaling = 0.125f;      // HD^-0.5 = 1/8

    // Q/K/V projections (scaling folded into Q epilogue)
    gemm_wmma_bf16<<<gD, blk, 0, stream>>>(h_bf, wq_bf, q_bias, nullptr, nullptr, q_bf,
                                           DMODEL, DMODEL, scaling, 0, 0, 0);
    gemm_wmma_bf16<<<gD, blk, 0, stream>>>(h_bf, wk_bf, nullptr, nullptr, nullptr, k_bf,
                                           DMODEL, DMODEL, 1.0f, 0, 0, 0);
    gemm_wmma_bf16<<<gD, blk, 0, stream>>>(h_bf, wv_bf, v_bias, nullptr, nullptr, v_bf,
                                           DMODEL, DMODEL, 1.0f, 0, 0, 0);

    // attention
    dim3 gA(MPAD / 16, NHEAD);         // 96 x 20, one wave each
    flash_attn_wmma<<<gA, 32, 0, stream>>>(q_bf, k_bf, v_bf, ctx_bf);

    // out projection + residual(x) -> hres (fp32)
    gemm_wmma_bf16<<<gD, blk, 0, stream>>>(ctx_bf, wo_bf, out_bias, x, hres, nullptr,
                                           DMODEL, DMODEL, 1.0f, 0, S_LEN, MPAD);

    // LN2
    layernorm_to_bf16<<<MPAD, 256, 0, stream>>>(hres, ln2_w, ln2_b, h2_bf);

    // FC1 + GELU -> bf16
    gemm_wmma_bf16<<<gF, blk, 0, stream>>>(h2_bf, wf1_bf, fc1_bias, nullptr, nullptr, fc1_bf,
                                           DMODEL, FFN_DIM, 1.0f, 1, 0, 0);
    // FC2 + residual(hres) -> d_out (fp32, rows < S)
    gemm_wmma_bf16<<<gD, blk, 0, stream>>>(fc1_bf, wf2_bf, fc2_bias, hres, out, nullptr,
                                           FFN_DIM, DMODEL, 1.0f, 0, S_LEN, S_LEN);
}
